// Generator_53876069761253
// MI455X (gfx1250) — matrix operand order repaired and checked by reading, not compiled, not run on hardware
//
#include <hip/hip_runtime.h>
#include <hip/hip_bf16.h>
#include <cstdint>

#define DIMD 128
#define TLEN 20
#define BSZ  1024
#define NGRP 8
#define NHEAD 8

typedef __attribute__((ext_vector_type(16))) _Float16 v16h;
typedef __attribute__((ext_vector_type(8)))  _Float16 v8h;
typedef __attribute__((ext_vector_type(8)))  float    v8f;

// ---------------------------------------------------------------------------
// FAST WMMA GEMM: C[M,N] = act(A[M,K] @ W[K,N] + bias)
// Requires M%64==0, N%128==0, K%32==0.
// Block: 256 threads = 8 waves; block tile 64(M) x 128(N).
// Wave (wy=w>>2 in 0..1, wx=w&3 in 0..3) computes a 32x32 strip:
// 2 A-frags x 2 B-frags = 4 WMMAs per k-step.
// LDS: sA[m][k] 64x32 f16, sBt[n][k] 128x32 f16 (transposed).
// Fragment loads are 2x ds_load_b128 each.
// ---------------------------------------------------------------------------
template<bool RELU>
__global__ __launch_bounds__(256) void k_gemm_fast(
    const float* __restrict__ A, const float* __restrict__ W,
    const float* __restrict__ bias, float* __restrict__ C,
    int M, int N, int K)
{
  __shared__ _Float16 sA [64 * 32];
  __shared__ _Float16 sBt[128 * 32];
  const int tid  = threadIdx.x;
  const int lane = tid & 31;
  const int wave = tid >> 5;
  const int wy   = wave >> 2;          // 0..1  (M sub-strip of 32)
  const int wx   = wave & 3;           // 0..3  (N sub-strip of 32)
  const int m0   = blockIdx.y * 64;
  const int n0   = blockIdx.x * 128;
  const int half = lane >> 4;
  const int l15  = lane & 15;

  // staging coordinates
  const int ar = tid >> 2;             // A row 0..63
  const int ak = (tid & 3) * 8;        // A k-offset 0,8,16,24
  const int bn = tid & 127;            // B column 0..127
  const int bk = (tid >> 7) * 16;      // B k-offset 0 or 16

  v8f acc00 = {}, acc01 = {}, acc10 = {}, acc11 = {};

  for (int k0 = 0; k0 < K; k0 += 32) {
    // ---- stage A: 2x float4 -> v8h -> ds_store_b128 ----
    {
      const float4* ap = (const float4*)(A + (size_t)(m0 + ar) * K + k0 + ak);
      float4 f0 = ap[0], f1 = ap[1];
      v8h p;
      p[0]=(_Float16)f0.x; p[1]=(_Float16)f0.y; p[2]=(_Float16)f0.z; p[3]=(_Float16)f0.w;
      p[4]=(_Float16)f1.x; p[5]=(_Float16)f1.y; p[6]=(_Float16)f1.z; p[7]=(_Float16)f1.w;
      *(v8h*)(sA + ar * 32 + ak) = p;
    }
    // ---- stage B transposed: 16 k-strided dwords (lane-coalesced in n) ----
    {
      const float* wp = W + (size_t)(k0 + bk) * N + n0 + bn;
      v8h p0, p1;
#pragma unroll
      for (int q = 0; q < 8; ++q) p0[q] = (_Float16)wp[(size_t)q * N];
#pragma unroll
      for (int q = 0; q < 8; ++q) p1[q] = (_Float16)wp[(size_t)(q + 8) * N];
      *(v8h*)(sBt + bn * 32 + bk)     = p0;
      *(v8h*)(sBt + bn * 32 + bk + 8) = p1;
    }
    if (k0 + 32 < K)
      __builtin_prefetch(A + (size_t)(m0 + ar) * K + k0 + 32 + ak, 0, 1);
    __syncthreads();

    // ---- fragments (ISA 7.12.2 layouts) ----
    const int mr0 = (wy * 32 + l15) * 32;
    v8h a00 = *(const v8h*)(sA + mr0 + half * 8);
    v8h a01 = *(const v8h*)(sA + mr0 + 16 + half * 8);
    v16h af0 = __builtin_shufflevector(a00, a01, 0,1,2,3,4,5,6,7,8,9,10,11,12,13,14,15);

    const int mr1 = (wy * 32 + 16 + l15) * 32;
    v8h a10 = *(const v8h*)(sA + mr1 + half * 8);
    v8h a11 = *(const v8h*)(sA + mr1 + 16 + half * 8);
    v16h af1 = __builtin_shufflevector(a10, a11, 0,1,2,3,4,5,6,7,8,9,10,11,12,13,14,15);

    const int nc0 = (wx * 32 + l15) * 32 + half * 8;
    v8h b00 = *(const v8h*)(sBt + nc0);
    v8h b01 = *(const v8h*)(sBt + nc0 + 16);
    v16h bf0 = __builtin_shufflevector(b00, b01, 0,1,2,3,4,5,6,7,8,9,10,11,12,13,14,15);

    const int nc1 = (wx * 32 + 16 + l15) * 32 + half * 8;
    v8h b10 = *(const v8h*)(sBt + nc1);
    v8h b11 = *(const v8h*)(sBt + nc1 + 16);
    v16h bf1 = __builtin_shufflevector(b10, b11, 0,1,2,3,4,5,6,7,8,9,10,11,12,13,14,15);

    acc00 = __builtin_amdgcn_wmma_f32_16x16x32_f16(false, af0, false, bf0,
                                                   (short)0, acc00, false, false);
    acc01 = __builtin_amdgcn_wmma_f32_16x16x32_f16(false, af0, false, bf1,
                                                   (short)0, acc01, false, false);
    acc10 = __builtin_amdgcn_wmma_f32_16x16x32_f16(false, af1, false, bf0,
                                                   (short)0, acc10, false, false);
    acc11 = __builtin_amdgcn_wmma_f32_16x16x32_f16(false, af1, false, bf1,
                                                   (short)0, acc11, false, false);
    __syncthreads();
  }

  // ---- epilogue: c[r] -> (m = base + r + 8*half, n = base + l15) ----
  const int na = n0 + wx * 32 + l15;
  const int nb = na + 16;
  const float bva = bias[na];
  const float bvb = bias[nb];
  const int ma0 = m0 + wy * 32 + 8 * half;
  const int ma1 = ma0 + 16;
#pragma unroll
  for (int r = 0; r < 8; ++r) {
    float v00 = acc00[r] + bva;
    float v01 = acc01[r] + bvb;
    float v10 = acc10[r] + bva;
    float v11 = acc11[r] + bvb;
    if (RELU) {
      v00 = fmaxf(v00, 0.f); v01 = fmaxf(v01, 0.f);
      v10 = fmaxf(v10, 0.f); v11 = fmaxf(v11, 0.f);
    }
    C[(size_t)(ma0 + r) * N + na] = v00;
    C[(size_t)(ma0 + r) * N + nb] = v01;
    C[(size_t)(ma1 + r) * N + na] = v10;
    C[(size_t)(ma1 + r) * N + nb] = v11;
  }
}

// ---------------------------------------------------------------------------
// Guarded WMMA GEMM for ragged shapes (K=176, N=2). Rarely used; slow is OK.
// ---------------------------------------------------------------------------
template<bool RELU>
__global__ __launch_bounds__(256) void k_gemm(
    const float* __restrict__ A, const float* __restrict__ W,
    const float* __restrict__ bias, float* __restrict__ C,
    int M, int N, int K)
{
  __shared__ _Float16 sA[64 * 32];
  __shared__ _Float16 sB[32 * 32];
  const int tid  = threadIdx.x;
  const int lane = tid & 31;
  const int wave = tid >> 5;
  const int wy   = wave >> 1;
  const int wx   = wave & 1;
  const int m0   = blockIdx.y * 64;
  const int n0   = blockIdx.x * 32;
  const int half = lane >> 4;
  const int l15  = lane & 15;

  v8f acc = {};

  for (int k0 = 0; k0 < K; k0 += 32) {
#pragma unroll
    for (int i = 0; i < 8; ++i) {
      int idx = tid + i * 256;
      int r = idx >> 5, c = idx & 31;
      int gm = m0 + r, gk = k0 + c;
      float v = (gm < M && gk < K) ? A[(size_t)gm * K + gk] : 0.f;
      sA[idx] = (_Float16)v;
    }
#pragma unroll
    for (int i = 0; i < 4; ++i) {
      int idx = tid + i * 256;
      int r = idx >> 5, c = idx & 31;
      int gk = k0 + r, gn = n0 + c;
      float v = (gk < K && gn < N) ? W[(size_t)gk * N + gn] : 0.f;
      sB[idx] = (_Float16)v;
    }
    __syncthreads();

    v16h af, bf;
    const int mrow = wy * 16 + l15;
#pragma unroll
    for (int j = 0; j < 8; ++j) {
      af[j]     = sA[mrow * 32 + half * 8 + j];
      af[j + 8] = sA[mrow * 32 + 16 + half * 8 + j];
    }
    const int ncol = wx * 16 + l15;
#pragma unroll
    for (int j = 0; j < 8; ++j) {
      bf[j]     = sB[(half * 8 + j) * 32 + ncol];
      bf[j + 8] = sB[(16 + half * 8 + j) * 32 + ncol];
    }

    acc = __builtin_amdgcn_wmma_f32_16x16x32_f16(false, af, false, bf,
                                                 (short)0, acc, false, false);
    __syncthreads();
  }

  const int n = n0 + wx * 16 + l15;
  const float bv = (n < N) ? bias[n] : 0.f;
#pragma unroll
  for (int r = 0; r < 8; ++r) {
    int m = m0 + wy * 16 + r + 8 * half;
    float v = acc[r] + bv;
    if (RELU) v = fmaxf(v, 0.f);
    if (m < M && n < N) C[(size_t)m * N + n] = v;
  }
}

// ---------------------------------------------------------------------------
// Attention: one wave per (sequence, head). qkv rows = seq*S + pos, 384 wide.
// ---------------------------------------------------------------------------
template<int S, bool CAUSAL>
__global__ __launch_bounds__(64) void k_attn(
    const float* __restrict__ qkv, float* __restrict__ out, int nseq)
{
  const int gwave = (blockIdx.x * blockDim.x + threadIdx.x) >> 5;
  const int lane  = threadIdx.x & 31;
  const int seq   = gwave >> 3;
  const int h     = gwave & 7;
  if (seq >= nseq) return;
  const float* base = qkv + (size_t)seq * S * 384;
  const int q = lane;
  if (q >= S) return;

  float qv[16];
#pragma unroll
  for (int j = 0; j < 16; ++j) qv[j] = base[q * 384 + h * 16 + j];

  float sc[S];
#pragma unroll
  for (int k = 0; k < S; ++k) {
    float s = 0.f;
#pragma unroll
    for (int j = 0; j < 16; ++j) s += qv[j] * base[k * 384 + 128 + h * 16 + j];
    s *= 0.25f;
    if (CAUSAL && k > q) s += -1.0e9f;
    sc[k] = s;
  }
  float mx = sc[0];
#pragma unroll
  for (int k = 1; k < S; ++k) mx = fmaxf(mx, sc[k]);
  float sum = 0.f;
#pragma unroll
  for (int k = 0; k < S; ++k) { sc[k] = __expf(sc[k] - mx); sum += sc[k]; }
  const float inv = 1.f / sum;

  float o[16];
#pragma unroll
  for (int j = 0; j < 16; ++j) o[j] = 0.f;
#pragma unroll
  for (int k = 0; k < S; ++k) {
    float p = sc[k] * inv;
#pragma unroll
    for (int j = 0; j < 16; ++j) o[j] += p * base[k * 384 + 256 + h * 16 + j];
  }
#pragma unroll
  for (int j = 0; j < 16; ++j)
    out[(size_t)(seq * S + q) * 128 + h * 16 + j] = o[j];
}

// ---------------------------------------------------------------------------
// x = LayerNorm(x + h) * g + b  (row width 128, one wave per row)
// ---------------------------------------------------------------------------
__global__ __launch_bounds__(128) void k_ln_res(
    float* __restrict__ x, const float* __restrict__ h,
    const float* __restrict__ g, const float* __restrict__ b, int M)
{
  const int row  = (blockIdx.x * blockDim.x + threadIdx.x) >> 5;
  const int lane = threadIdx.x & 31;
  if (row >= M) return;
  float v[4];
  float s = 0.f;
  const float4 xr = *(const float4*)(x + (size_t)row * 128 + lane * 4);
  const float4 hr = *(const float4*)(h + (size_t)row * 128 + lane * 4);
  v[0] = xr.x + hr.x; v[1] = xr.y + hr.y; v[2] = xr.z + hr.z; v[3] = xr.w + hr.w;
  s = v[0] + v[1] + v[2] + v[3];
#pragma unroll
  for (int off = 16; off > 0; off >>= 1) s += __shfl_xor(s, off, 32);
  const float mu = s * (1.f / 128.f);
  float var = 0.f;
#pragma unroll
  for (int j = 0; j < 4; ++j) { float d = v[j] - mu; var += d * d; }
#pragma unroll
  for (int off = 16; off > 0; off >>= 1) var += __shfl_xor(var, off, 32);
  var *= (1.f / 128.f);
  const float inv = rsqrtf(var + 1e-5f);
  const float4 gr = *(const float4*)(g + lane * 4);
  const float4 br = *(const float4*)(b + lane * 4);
  float4 o;
  o.x = (v[0] - mu) * inv * gr.x + br.x;
  o.y = (v[1] - mu) * inv * gr.y + br.y;
  o.z = (v[2] - mu) * inv * gr.z + br.z;
  o.w = (v[3] - mu) * inv * gr.w + br.w;
  *(float4*)(x + (size_t)row * 128 + lane * 4) = o;
}

// ---------------------------------------------------------------------------
// t_in = relu([rel_traj|target] @ Wcat + bcat) + pos_encoding(t)
// ---------------------------------------------------------------------------
__global__ void k_embed(const float* __restrict__ rel_traj,
                        const float* __restrict__ target,
                        const float* __restrict__ Wcat,
                        const float* __restrict__ bcat,
                        float* __restrict__ out)
{
  const int idx = blockIdx.x * blockDim.x + threadIdx.x;
  if (idx >= BSZ * TLEN * 128) return;
  const int c = idx & 127;
  const int tok = idx >> 7;
  const int t = tok % TLEN;
  const float i0 = rel_traj[tok * 2 + 0], i1 = rel_traj[tok * 2 + 1];
  const float i2 = target[tok * 2 + 0],   i3 = target[tok * 2 + 1];
  float v = i0 * Wcat[c] + i1 * Wcat[128 + c] + i2 * Wcat[256 + c]
          + i3 * Wcat[384 + c] + bcat[c];
  v = fmaxf(v, 0.f);
  const int i = c >> 1;
  const float div = __powf(10000.f, (float)(2 * i) * (1.f / 128.f));
  const float ang = (float)t / div;
  v += (c & 1) ? __cosf(ang) : __sinf(ang);
  out[idx] = v;
}

// ---------------------------------------------------------------------------
// Space-encoder input for time slice t (8192 x 256 concat tile)
// ---------------------------------------------------------------------------
__global__ void k_space_in(const float* __restrict__ traj,
                           const float* __restrict__ tout,
                           const float* __restrict__ Wpos,
                           const float* __restrict__ bpos,
                           float* __restrict__ cat, int t)
{
  const int idx = blockIdx.x * blockDim.x + threadIdx.x;
  if (idx >= 8192 * 256) return;
  const int c = idx & 255;
  const int row = idx >> 8;
  const int j = row & 7;
  const int i = (row >> 3) & 7;
  const int g = row >> 6;
  if (c < 128) {
    const int bi = g * 8 + i, bj = g * 8 + j;
    const float r0 = traj[(bi * TLEN + t) * 2 + 0] - traj[(bj * TLEN + t) * 2 + 0];
    const float r1 = traj[(bi * TLEN + t) * 2 + 1] - traj[(bj * TLEN + t) * 2 + 1];
    cat[idx] = fmaxf(r0 * Wpos[c] + r1 * Wpos[128 + c] + bpos[c], 0.f);
  } else {
    const int bj = g * 8 + j;
    cat[idx] = tout[((size_t)bj * TLEN + t) * 128 + (c - 128)];
  }
}

// enc chunk diag (j == i) -> s_out[(b*T + t)*128 + c], b = g*8+i
__global__ void k_diag(const float* __restrict__ enc,
                       float* __restrict__ sdiag, int t)
{
  const int idx = blockIdx.x * blockDim.x + threadIdx.x;
  if (idx >= 1024 * 128) return;
  const int c = idx & 127;
  const int b = idx >> 7;
  const int g = b >> 3, i = b & 7;
  const int row = g * 64 + i * 9;
  sdiag[((size_t)b * TLEN + t) * 128 + c] = enc[(size_t)row * 128 + c];
}

// f_in = [s_out(128) | noise(16) | relu(col*Wcol + bcol)(32)]  (20480 x 176)
__global__ void k_fin(const float* __restrict__ sdiag,
                      const float* __restrict__ noise,
                      const float* __restrict__ col_label,
                      const float* __restrict__ Wcol,
                      const float* __restrict__ bcol,
                      float* __restrict__ fin)
{
  const int idx = blockIdx.x * blockDim.x + threadIdx.x;
  if (idx >= BSZ * TLEN * 176) return;
  const int c = idx % 176;
  const int tok = idx / 176;
  float v;
  if (c < 128)       v = sdiag[(size_t)tok * 128 + c];
  else if (c < 144)  v = noise[(size_t)tok * 16 + (c - 128)];
  else {
    const int cc = c - 144;
    v = fmaxf(col_label[tok] * Wcol[cc] + bcol[cc], 0.f);
  }
  fin[idx] = v;
}

// ---------------------------------------------------------------------------
extern "C" void kernel_launch(void* const* d_in, const int* in_sizes, int n_in,
                              void* d_out, int out_size, void* d_ws, size_t ws_size,
                              hipStream_t stream)
{
  (void)in_sizes; (void)n_in; (void)out_size; (void)ws_size;
  const float* traj      = (const float*)d_in[0];
  const float* rel_traj  = (const float*)d_in[1];
  const float* target    = (const float*)d_in[2];
  const float* col_label = (const float*)d_in[3];
  const float* noise     = (const float*)d_in[4];
  const float* Wcat = (const float*)d_in[5];
  const float* bcat = (const float*)d_in[6];
  const float* Wpos = (const float*)d_in[7];
  const float* bpos = (const float*)d_in[8];
  const float* Whid = (const float*)d_in[9];
  const float* bhid = (const float*)d_in[10];
  const float* Wcol = (const float*)d_in[11];
  const float* bcol = (const float*)d_in[12];
  const float* Wf1  = (const float*)d_in[13];
  const float* bf1  = (const float*)d_in[14];
  const float* Wf2  = (const float*)d_in[15];
  const float* bf2  = (const float*)d_in[16];

  struct Layer { const float *Wqkv,*bqkv,*Wo,*bo,*g1,*e1,*W1,*b1,*W2,*b2,*g2,*e2; };
  auto layer = [&](int base) {
    Layer L;
    L.Wqkv=(const float*)d_in[base+0];  L.bqkv=(const float*)d_in[base+1];
    L.Wo  =(const float*)d_in[base+2];  L.bo  =(const float*)d_in[base+3];
    L.g1  =(const float*)d_in[base+4];  L.e1  =(const float*)d_in[base+5];
    L.W1  =(const float*)d_in[base+6];  L.b1  =(const float*)d_in[base+7];
    L.W2  =(const float*)d_in[base+8];  L.b2  =(const float*)d_in[base+9];
    L.g2  =(const float*)d_in[base+10]; L.e2  =(const float*)d_in[base+11];
    return L;
  };
  const Layer tl0 = layer(17), tl1 = layer(29), sl0 = layer(41), sl1 = layer(53);

  const size_t NT1 = (size_t)BSZ * TLEN;
  float* ws = (float*)d_ws;
  float* bufTout = ws; ws += NT1 * 128;
  float* bufQKV  = ws; ws += NT1 * 384;
  float* bufAttn = ws; ws += NT1 * 128;
  float* bufH    = ws; ws += NT1 * 512;
  float* bufTmp  = ws; ws += NT1 * 128;
  float* bufX    = ws; ws += NT1 * 128;
  float* bufS    = ws; ws += NT1 * 128;
  float* bufCat  = ws; ws += NT1 * 176;

  auto gemm = [&](const float* A, const float* W, const float* b, float* C,
                  int M, int N, int K, bool relu) {
    if ((M % 64 == 0) && (N % 128 == 0) && (K % 32 == 0)) {
      dim3 grid(N / 128, M / 64), blk(256);
      if (relu) k_gemm_fast<true ><<<grid, blk, 0, stream>>>(A, W, b, C, M, N, K);
      else      k_gemm_fast<false><<<grid, blk, 0, stream>>>(A, W, b, C, M, N, K);
    } else {
      dim3 grid((N + 31) / 32, (M + 63) / 64), blk(256);
      if (relu) k_gemm<true ><<<grid, blk, 0, stream>>>(A, W, b, C, M, N, K);
      else      k_gemm<false><<<grid, blk, 0, stream>>>(A, W, b, C, M, N, K);
    }
  };
  auto ln = [&](float* X, const float* H, const float* g, const float* b, int M) {
    k_ln_res<<<(M + 3) / 4, 128, 0, stream>>>(X, H, g, b, M);
  };
  auto enc_layer = [&](float* X, const Layer& L, int M, int nseq, bool time_mode) {
    gemm(X, L.Wqkv, L.bqkv, bufQKV, M, 384, 128, false);
    if (time_mode)
      k_attn<TLEN, true ><<<(nseq * NHEAD * 32 + 63) / 64, 64, 0, stream>>>(bufQKV, bufAttn, nseq);
    else
      k_attn<NGRP, false><<<(nseq * NHEAD * 32 + 63) / 64, 64, 0, stream>>>(bufQKV, bufAttn, nseq);
    gemm(bufAttn, L.Wo, L.bo, bufTmp, M, 128, 128, false);
    ln(X, bufTmp, L.g1, L.e1, M);
    gemm(X, L.W1, L.b1, bufH, M, 512, 128, true);
    gemm(bufH, L.W2, L.b2, bufTmp, M, 128, 512, false);
    ln(X, bufTmp, L.g2, L.e2, M);
  };

  // ---- time encoder (B=1024 seqs of T=20, causal) ----
  k_embed<<<((int)(NT1 * 128) + 255) / 256, 256, 0, stream>>>(rel_traj, target, Wcat, bcat, bufTout);
  enc_layer(bufTout, tl0, (int)NT1, BSZ, true);
  enc_layer(bufTout, tl1, (int)NT1, BSZ, true);

  // ---- space encoder, chunked per time step (8192 tokens, 1024 seqs of 8) ----
  for (int t = 0; t < TLEN; ++t) {
    k_space_in<<<(8192 * 256 + 255) / 256, 256, 0, stream>>>(traj, bufTout, Wpos, bpos, bufCat, t);
    gemm(bufCat, Whid, bhid, bufX, 8192, 128, 256, true);
    enc_layer(bufX, sl0, 8192, 1024, false);
    enc_layer(bufX, sl1, 8192, 1024, false);
    k_diag<<<(1024 * 128 + 255) / 256, 256, 0, stream>>>(bufX, bufS, t);
  }

  // ---- final MLP ----
  k_fin<<<((int)(NT1 * 176) + 255) / 256, 256, 0, stream>>>(bufS, noise, col_label, Wcol, bcol, bufCat);
  gemm(bufCat, Wf1, bf1, bufTmp, (int)NT1, 128, 176, true);
  gemm(bufTmp, Wf2, bf2, (float*)d_out, (int)NT1, 2, 128, false);
}
